// OrientationLearner_32160715112613
// MI455X (gfx1250) — matrix operand order repaired and checked by reading, not compiled, not run on hardware
//
#include <hip/hip_runtime.h>
#include <math.h>

#define N_NODES 50000
#define N_EDGES 800000
#define HIDDEN  128
#define EDGE_NF 32
#define K1      288   // 2*HIDDEN + EDGE_NF
#define EPSF    1e-12f

typedef __attribute__((ext_vector_type(16))) __bf16    v16bf;
typedef __attribute__((ext_vector_type(8)))  float     v8f;
typedef __attribute__((ext_vector_type(4)))  unsigned  v4u;
typedef __attribute__((ext_vector_type(2)))  unsigned  v2u;

union FragBF { v16bf v; v4u q[2]; };

// Fragment-packed storage: one 32-K block = 64 v4u:
//   dword offset(blk, half, lane, q) = ((blk*2 + half)*32 + lane)*4 + q , j = half*4+q
// A-mapping (ISA 7.12.2, 16-bit A 16x32): element(m, kb): hi=(kb>>3)&1,
//   j = ((kb>>4)<<2) + ((kb&7)>>1), lane = m + 16*hi, dword holds (k, k+1)
// B-mapping (dual, 32x16):            element(kb, n): hi=(kb>>3)&1, j=((kb>>4)<<2)+((kb&7)>>1),
//   lane = n + 16*hi? no -> lane = (n&15) + 16*hi with n per ntile; dword holds (k, k+1)

__device__ __forceinline__ unsigned short f2bf(float f) {
  unsigned u = __float_as_uint(f);
  return (unsigned short)((u + 0x7FFFu + ((u >> 16) & 1u)) >> 16);  // RNE
}
__device__ __forceinline__ unsigned pack2bf(float a, float b) {
  return (unsigned)f2bf(a) | ((unsigned)f2bf(b) << 16);
}
__device__ __forceinline__ float silu(float x) { return x / (1.0f + __expf(-x)); }

// store 4 consecutive k-elements (k0 % 4 == 0) of row m into an A-fragment-packed tile
__device__ __forceinline__ void store4bf_sw(unsigned* tile, int m, int k0, float4 v) {
  int ks = k0 >> 5, kb = k0 & 31;
  int hi = (kb >> 3) & 1;
  int j  = ((kb >> 4) << 2) + ((kb & 7) >> 1);          // even
  int lane_f = m + (hi << 4);
  unsigned* p = tile + (((ks * 2 + (j >> 2)) * 32 + lane_f) << 2) + (j & 3);
  v2u t = { pack2bf(v.x, v.y), pack2bf(v.z, v.w) };      // (j&3) in {0,2} -> 8B aligned
  *(v2u*)p = t;
}

// fragment load: two ds_load_b128
__device__ __forceinline__ v16bf load_frag_blk(const v4u* buf, int blk, int lane) {
  FragBF f;
  f.q[0] = buf[(blk * 2) * 32 + lane];
  f.q[1] = buf[(blk * 2 + 1) * 32 + lane];
  return f.v;
}

__global__ void ol_zero_kernel(float* p, int n) {
  int i = blockIdx.x * blockDim.x + threadIdx.x;
  if (i < n) p[i] = 0.0f;
}

// pack a row-major [K][128] f32 weight into B-fragment-packed bf16 dwords
__global__ void ol_pack_b_kernel(const float* __restrict__ src, unsigned* __restrict__ dst, int K) {
  int i = blockIdx.x * blockDim.x + threadIdx.x;   // output dword index
  if (i >= K * 64) return;
  int q    = i & 3;
  int lane = (i >> 2) & 31;
  int half = (i >> 7) & 1;
  int t    = (i >> 8) & 7;
  int ks   = i >> 11;
  int n  = (lane & 15) + t * 16;
  int hi = lane >> 4;
  int j  = half * 4 + q;
  int k  = ks * 32 + (j >> 2) * 16 + hi * 8 + 2 * (j & 3);
  dst[i] = pack2bf(src[k * HIDDEN + n], src[(k + 1) * HIDDEN + n]);
}

// ---------------- Edge MLP + scatter ----------------
// grid = N_EDGES/128, 256 threads (8 waves), each wave owns a 16-edge M-tile.
__global__ void __launch_bounds__(256) ol_edge_kernel(
    const float* __restrict__ h, const int* __restrict__ row, const int* __restrict__ col,
    const float* __restrict__ ea,
    const unsigned* __restrict__ W1p, const float* __restrict__ b1,
    const unsigned* __restrict__ W2p, const float* __restrict__ b2,
    float* __restrict__ s, float* __restrict__ cnt)
{
  extern __shared__ unsigned smem_u[];
  unsigned* W1s = smem_u;                          // K1*64      = 18432 dw
  unsigned* W2s = W1s + K1 * 64;                   // HIDDEN*64  =  8192 dw
  unsigned* A1w = W2s + HIDDEN * 64;               // 8 * 2304 dw
  unsigned* A2w = A1w + 8 * (K1 / 32) * 256;       // 8 * 1024 dw
  int* ridx = (int*)(A2w + 8 * (HIDDEN / 32) * 256);   // 128
  int* cidx = ridx + 128;                              // 128

  int tid = threadIdx.x, wave = tid >> 5, lane = tid & 31;

  { // stage pre-packed weights verbatim
    const v4u* sa = (const v4u*)W1p; v4u* da = (v4u*)W1s;
    for (int i = tid; i < K1 * 16; i += 256) da[i] = sa[i];
    const v4u* sb = (const v4u*)W2p; v4u* db = (v4u*)W2s;
    for (int i = tid; i < HIDDEN * 16; i += 256) db[i] = sb[i];
  }

  int ebase = blockIdx.x * 128 + wave * 16;
  if (lane < 16) {
    ridx[wave * 16 + lane] = row[ebase + lane];
    cidx[wave * 16 + lane] = col[ebase + lane];
  }

  // stage gathered features directly in A-fragment layout (conflict-free pattern):
  // lane owns row m = lane&15, walks chunks; hi half of wave takes odd chunks.
  int mm = lane & 15, hil = lane >> 4;
  unsigned* tileA = A1w + wave * (K1 / 32) * 256;
  {
    int r = ridx[wave * 16 + mm];
    int c = cidx[wave * 16 + mm];
    int e = ebase + mm;
    for (int cc = 0; cc < 16; ++cc) {
      int k0 = cc * 8 + hil * 4;
      float4 v = *(const float4*)(h + (size_t)r * HIDDEN + k0);
      store4bf_sw(tileA, mm, k0, v);
    }
    for (int cc = 0; cc < 16; ++cc) {
      int k0 = cc * 8 + hil * 4;
      float4 v = *(const float4*)(h + (size_t)c * HIDDEN + k0);
      store4bf_sw(tileA, mm, 128 + k0, v);
    }
    for (int cc = 0; cc < 4; ++cc) {
      int k0 = cc * 8 + hil * 4;
      float4 v = *(const float4*)(ea + (size_t)e * EDGE_NF + k0);
      store4bf_sw(tileA, mm, 256 + k0, v);
    }
  }
  __syncthreads();

  int nl = lane & 15, hi = lane >> 4;
  v8f z = {0.f, 0.f, 0.f, 0.f, 0.f, 0.f, 0.f, 0.f};
  const v4u* Aq = (const v4u*)tileA;
  const v4u* W1q = (const v4u*)W1s;
  const v4u* W2q = (const v4u*)W2s;

  // ---- GEMM1: [16,288] x [288,128] ----
  v8f acc[8];
#pragma unroll
  for (int t = 0; t < 8; ++t) acc[t] = z;
  for (int ks = 0; ks < K1 / 32; ++ks) {
    v16bf a = load_frag_blk(Aq, ks, lane);
#pragma unroll
    for (int t = 0; t < 8; ++t) {
      v16bf b = load_frag_blk(W1q, ks * 8 + t, lane);
      acc[t] = __builtin_amdgcn_wmma_f32_16x16x32_bf16(false, a, false, b, (short)0, acc[t], false, false);
    }
  }

  // bias + SiLU -> A2 tile, written straight into A-fragment layout
  unsigned short* A2s = (unsigned short*)(A2w + wave * (HIDDEN / 32) * 256);
#pragma unroll
  for (int t = 0; t < 8; ++t) {
    float bb = b1[t * 16 + nl];
#pragma unroll
    for (int r = 0; r < 8; ++r) {
      float x = silu(acc[t][r] + bb);
      int n = t * 16 + nl, m2 = r + 8 * hi;
      int ks2 = n >> 5, kb = n & 31;
      int hia = (kb >> 3) & 1;
      int j = ((kb >> 4) << 2) + ((kb & 7) >> 1);
      int dw = ((ks2 * 2 + (j >> 2)) * 32 + (m2 + (hia << 4))) * 4 + (j & 3);
      A2s[dw * 2 + (n & 1)] = f2bf(x);
    }
  }

  // ---- GEMM2: [16,128] x [128,128] ----
  const v4u* A2q = (const v4u*)(A2w + wave * 1024);
#pragma unroll
  for (int t = 0; t < 8; ++t) acc[t] = z;
  for (int ks = 0; ks < HIDDEN / 32; ++ks) {
    v16bf a = load_frag_blk(A2q, ks, lane);
#pragma unroll
    for (int t = 0; t < 8; ++t) {
      v16bf b = load_frag_blk(W2q, ks * 8 + t, lane);
      acc[t] = __builtin_amdgcn_wmma_f32_16x16x32_bf16(false, a, false, b, (short)0, acc[t], false, false);
    }
  }

  // scatter-add (accumulators L2-resident)
  if (lane < 16)
    __hip_atomic_fetch_add(&cnt[ridx[wave * 16 + lane]], 1.0f,
                           __ATOMIC_RELAXED, __HIP_MEMORY_SCOPE_AGENT);
#pragma unroll
  for (int t = 0; t < 8; ++t) {
    float bb = b2[t * 16 + nl];
#pragma unroll
    for (int r = 0; r < 8; ++r) {
      float x = silu(acc[t][r] + bb);
      int rn = ridx[wave * 16 + (r + 8 * hi)];
      __hip_atomic_fetch_add(&s[(size_t)rn * HIDDEN + t * 16 + nl], x,
                             __ATOMIC_RELAXED, __HIP_MEMORY_SCOPE_AGENT);
    }
  }
}

// ---------------- Node MLPs + Gram-Schmidt ----------------
__device__ __forceinline__ void mlp_head(
    const v4u* Aq, const v4u* Wq, const float* b1v,
    const float* w2, const float* b2v, float* act, int lane,
    float& o0, float& o1, float& o2)
{
  v8f z = {0.f, 0.f, 0.f, 0.f, 0.f, 0.f, 0.f, 0.f};
  v8f acc[8];
#pragma unroll
  for (int t = 0; t < 8; ++t) acc[t] = z;
  for (int ks = 0; ks < HIDDEN / 32; ++ks) {
    v16bf a = load_frag_blk(Aq, ks, lane);
#pragma unroll
    for (int t = 0; t < 8; ++t) {
      v16bf b = load_frag_blk(Wq, ks * 8 + t, lane);
      acc[t] = __builtin_amdgcn_wmma_f32_16x16x32_bf16(false, a, false, b, (short)0, acc[t], false, false);
    }
  }
  int nl = lane & 15, hi = lane >> 4;
#pragma unroll
  for (int t = 0; t < 8; ++t) {
    float bb = b1v[t * 16 + nl];
#pragma unroll
    for (int r = 0; r < 8; ++r)
      act[(r + 8 * hi) * HIDDEN + t * 16 + nl] = silu(acc[t][r] + bb);
  }
  // same-wave LDS ops are in-order; lanes 0-15 reduce their own node row
  o0 = o1 = o2 = 0.f;
  if (lane < 16) {
    const float4* a4 = (const float4*)(act + lane * HIDDEN);
    float s0 = b2v[0], s1 = b2v[1], s2 = b2v[2];
    for (int i = 0; i < HIDDEN / 4; ++i) {
      float4 v = a4[i];
      const float* w = w2 + i * 12;
      s0 += v.x * w[0] + v.y * w[3] + v.z * w[6] + v.w * w[9];
      s1 += v.x * w[1] + v.y * w[4] + v.z * w[7] + v.w * w[10];
      s2 += v.x * w[2] + v.y * w[5] + v.z * w[8] + v.w * w[11];
    }
    o0 = s0; o1 = s1; o2 = s2;
  }
}

__global__ void __launch_bounds__(256) ol_node_kernel(
    const float* __restrict__ s, const float* __restrict__ cnt,
    const unsigned* __restrict__ V1p, const float* __restrict__ v1b1,
    const float* __restrict__ v1w2, const float* __restrict__ v1b2,
    const unsigned* __restrict__ V2p, const float* __restrict__ v2b1,
    const float* __restrict__ v2w2, const float* __restrict__ v2b2,
    float* __restrict__ out, int nNodes)
{
  extern __shared__ unsigned smem_u[];
  unsigned* Wa = smem_u;                    // 8192 dw
  unsigned* Wb = Wa + HIDDEN * 64;          // 8192 dw
  unsigned* Aw = Wb + HIDDEN * 64;          // 8 * 1024 dw
  float* ACT = (float*)(Aw + 8 * 1024);     // 8 * 2048 f32

  int tid = threadIdx.x, wave = tid >> 5, lane = tid & 31;
  {
    const v4u* sa = (const v4u*)V1p; v4u* da = (v4u*)Wa;
    for (int i = tid; i < HIDDEN * 16; i += 256) da[i] = sa[i];
    const v4u* sb = (const v4u*)V2p; v4u* db = (v4u*)Wb;
    for (int i = tid; i < HIDDEN * 16; i += 256) db[i] = sb[i];
  }

  unsigned* at = Aw + wave * 1024;
  float* act = ACT + wave * 2048;
  int nb = (blockIdx.x * 8 + wave) * 16;
  int mm = lane & 15, hil = lane >> 4;
  {
    int node = nb + mm; if (node >= nNodes) node = nNodes - 1;
    float inv = 1.0f / fmaxf(cnt[node], 1.0f);       // scatter-mean
    for (int cc = 0; cc < 16; ++cc) {
      int k0 = cc * 8 + hil * 4;
      float4 v = *(const float4*)(s + (size_t)node * HIDDEN + k0);
      v.x *= inv; v.y *= inv; v.z *= inv; v.w *= inv;
      store4bf_sw(at, mm, k0, v);
    }
  }
  __syncthreads();

  float a0, a1, a2, c0, c1, c2;
  mlp_head((const v4u*)at, (const v4u*)Wa, v1b1, v1w2, v1b2, act, lane, a0, a1, a2);
  mlp_head((const v4u*)at, (const v4u*)Wb, v2b1, v2w2, v2b2, act, lane, c0, c1, c2);

  if (lane < 16) {
    int node = nb + lane;
    if (node < nNodes) {
      float n1 = fmaxf(sqrtf(a0 * a0 + a1 * a1 + a2 * a2), EPSF);
      float x0 = a0 / n1, x1 = a1 / n1, x2 = a2 / n1;
      float d = c0 * x0 + c1 * x1 + c2 * x2;
      float y0 = c0 - d * x0, y1 = c1 - d * x1, y2 = c2 - d * x2;
      float n2 = fmaxf(sqrtf(y0 * y0 + y1 * y1 + y2 * y2), EPSF);
      y0 /= n2; y1 /= n2; y2 /= n2;
      float z0 = x1 * y2 - x2 * y1;
      float z1 = x2 * y0 - x0 * y2;
      float z2 = x0 * y1 - x1 * y0;
      float* o = out + (size_t)node * 9;           // stack([v1n,v2n,v3], axis=-1)
      o[0] = x0; o[1] = y0; o[2] = z0;
      o[3] = x1; o[4] = y1; o[5] = z1;
      o[6] = x2; o[7] = y2; o[8] = z2;
    }
  }
}

extern "C" void kernel_launch(void* const* d_in, const int* in_sizes, int n_in,
                              void* d_out, int out_size, void* d_ws, size_t ws_size,
                              hipStream_t stream) {
  (void)in_sizes; (void)n_in; (void)out_size; (void)ws_size;
  const float* h     = (const float*)d_in[0];
  /* d_in[1] = x, unused by the reference */
  const int*   ei    = (const int*)d_in[2];
  const float* ea    = (const float*)d_in[3];
  const float* e_w1  = (const float*)d_in[4];
  const float* e_b1  = (const float*)d_in[5];
  const float* e_w2  = (const float*)d_in[6];
  const float* e_b2  = (const float*)d_in[7];
  const float* v1_w1 = (const float*)d_in[8];
  const float* v1_b1 = (const float*)d_in[9];
  const float* v1_w2 = (const float*)d_in[10];
  const float* v1_b2 = (const float*)d_in[11];
  const float* v2_w1 = (const float*)d_in[12];
  const float* v2_b1 = (const float*)d_in[13];
  const float* v2_w2 = (const float*)d_in[14];
  const float* v2_b2 = (const float*)d_in[15];

  float* s   = (float*)d_ws;                         // [N,128] scatter-sum
  float* cnt = s + (size_t)N_NODES * HIDDEN;         // [N]
  unsigned* W1p = (unsigned*)(cnt + N_NODES);        // K1*64 dw, fragment-packed bf16
  unsigned* W2p = W1p + K1 * 64;                     // HIDDEN*64 dw
  unsigned* V1p = W2p + HIDDEN * 64;
  unsigned* V2p = V1p + HIDDEN * 64;

  int ztot = N_NODES * HIDDEN + N_NODES;             // s and cnt contiguous
  ol_zero_kernel<<<(ztot + 255) / 256, 256, 0, stream>>>(s, ztot);
  ol_pack_b_kernel<<<(K1 * 64 + 255) / 256, 256, 0, stream>>>(e_w1, W1p, K1);
  ol_pack_b_kernel<<<(HIDDEN * 64 + 255) / 256, 256, 0, stream>>>(e_w2, W2p, HIDDEN);
  ol_pack_b_kernel<<<(HIDDEN * 64 + 255) / 256, 256, 0, stream>>>(v1_w1, V1p, HIDDEN);
  ol_pack_b_kernel<<<(HIDDEN * 64 + 255) / 256, 256, 0, stream>>>(v2_w1, V2p, HIDDEN);

  size_t smem1 = (size_t)(K1 * 64 + HIDDEN * 64 + 8 * (K1 / 32) * 256 + 8 * (HIDDEN / 32) * 256) * 4
               + 256 * sizeof(int);                  // 214016 B (< 320 KB/WGP)
  (void)hipFuncSetAttribute((const void*)ol_edge_kernel,
                            hipFuncAttributeMaxDynamicSharedMemorySize, (int)smem1);
  ol_edge_kernel<<<N_EDGES / 128, 256, smem1, stream>>>(
      h, ei, ei + N_EDGES, ea, W1p, e_b1, W2p, e_b2, s, cnt);

  size_t smem2 = (size_t)(3 * HIDDEN * 64) * 4 + (size_t)8 * 2048 * 4;   // 163840 B
  (void)hipFuncSetAttribute((const void*)ol_node_kernel,
                            hipFuncAttributeMaxDynamicSharedMemorySize, (int)smem2);
  int nblk = (N_NODES + 127) / 128;
  ol_node_kernel<<<nblk, 256, smem2, stream>>>(
      s, cnt, V1p, v1_b1, v1_w2, v1_b2, V2p, v2_b1, v2_w2, v2_b2,
      (float*)d_out, N_NODES);
}
